// CRF_86036784874155
// MI455X (gfx1250) — matrix operand order repaired and checked by reading, not compiled, not run on hardware
//
#include <hip/hip_runtime.h>
#include <hip/hip_bf16.h>

typedef __attribute__((ext_vector_type(16))) _Float16 v16h;
typedef __attribute__((ext_vector_type(8)))  float    v8f;

#define NUM_TAGS 64
#define SEQ_LEN  1024
#define BATCH    512
#define KSTRIDE  20      // dword stride of LDS transpose tile [k][m]; 80B keeps
                         // b128 stores 16B-aligned and b32 gathers conflict-free
#define LOG2E    1.4426950408889634f
#define LN2      0.6931471805599453f

// ---------------------------------------------------------------------------
// Normalizer: scaled forward algorithm in exp-domain via f16 WMMA.
// One wave per 16-row batch tile. alpha_hat kept in A-fragment layout:
//   lane L owns batch row (L&15); its 32 tag values are the lane's elements of
//   the two 16x32 f16 A fragments (K-halves merged with partner lane L^16).
// Per step: rowmax (in-lane tree + 1 shfl_xor16) -> A = exp(alpha - rm) ->
// 8 chained WMMAs vs resident Texp B fragments -> nw = v_log(acc)*ln2 + em in
// D layout -> LDS transpose tile -> branchless mask blend back into A layout.
// ---------------------------------------------------------------------------
__global__ __launch_bounds__(32)
void crf_normalizer_kernel(const float* __restrict__ emissions,
                           const int*   __restrict__ mask,
                           const float* __restrict__ startT,
                           const float* __restrict__ endT,
                           const float* __restrict__ transitions,
                           float* __restrict__ den_out)
{
    __shared__ float ldsT[NUM_TAGS * KSTRIDE];
    const int lane = threadIdx.x;
    const int col  = lane & 15;     // D-layout column / A-layout row
    const int hi   = lane >> 4;
    const int r0   = blockIdx.x * 16;
    const int row  = col;           // batch row owned by this lane (A layout)
    const int koff = hi ? 8 : 0;

    // lane's tag indices in WMMA A-fragment element order
    int kA[32];
#pragma unroll
    for (int e = 0; e < 8; ++e) { kA[e] = koff + e; kA[e + 8] = koff + 16 + e; }
#pragma unroll
    for (int e = 0; e < 16; ++e) kA[16 + e] = 32 + kA[e];

    // ---- B fragments: Texp = exp(transitions) in WMMA B layout (built once) ----
    v16h B[4][2];
#pragma unroll
    for (int j = 0; j < 4; ++j) {
#pragma unroll
        for (int h = 0; h < 2; ++h) {
            const int c = 16 * j + col;
            const int kbase = 32 * h + (hi ? 8 : 0);
#pragma unroll
            for (int e = 0; e < 16; ++e)
                B[j][h][e] = (_Float16)__builtin_amdgcn_exp2f(
                    transitions[(kbase + e + (e >= 8 ? 8 : 0)) * NUM_TAGS + c] * LOG2E);
        }
    }

    // ---- init: alpha_hat = start + emissions[0] (A layout), S = 0 ----
    float aA[32];
#pragma unroll
    for (int e = 0; e < 32; ++e)
        aA[e] = startT[kA[e]] + emissions[(size_t)(r0 + row) * NUM_TAGS + kA[e]];
    float S = 0.0f;

    for (int t = 1; t < SEQ_LEN; ++t) {
        // emission tile for this step, D layout; issued early to hide latency
        float em[4][8];
#pragma unroll
        for (int j = 0; j < 4; ++j) {
#pragma unroll
            for (int r = 0; r < 8; ++r)
                em[j][r] = emissions[((size_t)t * BATCH + r0 + 8 * hi + r) * NUM_TAGS
                                     + 16 * j + col];
        }
        if (t + 1 < SEQ_LEN) {  // prefetch next 4KB tile: one 128B line per lane
            const char* nxt = (const char*)(emissions
                              + ((size_t)(t + 1) * BATCH + r0) * NUM_TAGS);
            __builtin_prefetch(nxt + lane * 128, 0, 1);
        }
        const float mf = (float)mask[t * BATCH + r0 + row];

        // row max: 31-op in-lane tree + one xor16 to merge partner K-half
        float m16[16];
#pragma unroll
        for (int e = 0; e < 16; ++e) m16[e] = fmaxf(aA[e], aA[e + 16]);
#pragma unroll
        for (int e = 0; e < 8; ++e) m16[e] = fmaxf(m16[e], m16[e + 8]);
#pragma unroll
        for (int e = 0; e < 4; ++e) m16[e] = fmaxf(m16[e], m16[e + 4]);
        float rm = fmaxf(fmaxf(m16[0], m16[1]), fmaxf(m16[2], m16[3]));
        rm = fmaxf(rm, __shfl_xor(rm, 16, 32));

        // A fragments: exp(alpha - rm), raw v_exp (args <= 0, no denorm risk)
        v16h A0, A1;
#pragma unroll
        for (int e = 0; e < 16; ++e) {
            A0[e] = (_Float16)__builtin_amdgcn_exp2f((aA[e]      - rm) * LOG2E);
            A1[e] = (_Float16)__builtin_amdgcn_exp2f((aA[16 + e] - rm) * LOG2E);
        }

        asm volatile("s_wait_dscnt 0" ::: "memory");  // prev-iter gathers drained

        // 8 chained WMMAs; nw = v_log(acc)*ln2 + em (acc in [0.9, 71]: raw log safe)
#pragma unroll
        for (int j = 0; j < 4; ++j) {
            v8f acc = {};
            acc = __builtin_amdgcn_wmma_f32_16x16x32_f16(false, A0, false, B[j][0],
                                                         (short)0, acc, false, false);
            acc = __builtin_amdgcn_wmma_f32_16x16x32_f16(false, A1, false, B[j][1],
                                                         (short)0, acc, false, false);
            float4 lo, hiq;
            lo.x  = fmaf(__builtin_amdgcn_logf(acc[0]), LN2, em[j][0]);
            lo.y  = fmaf(__builtin_amdgcn_logf(acc[1]), LN2, em[j][1]);
            lo.z  = fmaf(__builtin_amdgcn_logf(acc[2]), LN2, em[j][2]);
            lo.w  = fmaf(__builtin_amdgcn_logf(acc[3]), LN2, em[j][3]);
            hiq.x = fmaf(__builtin_amdgcn_logf(acc[4]), LN2, em[j][4]);
            hiq.y = fmaf(__builtin_amdgcn_logf(acc[5]), LN2, em[j][5]);
            hiq.z = fmaf(__builtin_amdgcn_logf(acc[6]), LN2, em[j][6]);
            hiq.w = fmaf(__builtin_amdgcn_logf(acc[7]), LN2, em[j][7]);
            float4* p = (float4*)(ldsT + (16 * j + col) * KSTRIDE + 8 * hi);
            p[0] = lo;
            p[1] = hiq;
        }

        asm volatile("s_wait_dscnt 0" ::: "memory");  // transpose tile visible

        // gather back in A layout; branchless per-row mask blend
#pragma unroll
        for (int e = 0; e < 32; ++e) {
            const float nv = ldsT[kA[e] * KSTRIDE + row];
            aA[e] = fmaf(mf, nv - aA[e], aA[e]);
        }
        S = fmaf(mf, rm, S);
    }

    // ---- denominator: S + logsumexp(alpha_hat + end) (all in-lane + 2 shfl) ----
    float v[32];
#pragma unroll
    for (int e = 0; e < 32; ++e) v[e] = aA[e] + endT[kA[e]];
    float m16[16];
#pragma unroll
    for (int e = 0; e < 16; ++e) m16[e] = fmaxf(v[e], v[e + 16]);
#pragma unroll
    for (int e = 0; e < 8; ++e) m16[e] = fmaxf(m16[e], m16[e + 8]);
#pragma unroll
    for (int e = 0; e < 4; ++e) m16[e] = fmaxf(m16[e], m16[e + 4]);
    float rm = fmaxf(fmaxf(m16[0], m16[1]), fmaxf(m16[2], m16[3]));
    rm = fmaxf(rm, __shfl_xor(rm, 16, 32));
    float su = 0.0f;
#pragma unroll
    for (int e = 0; e < 32; ++e)
        su += __builtin_amdgcn_exp2f((v[e] - rm) * LOG2E);
    su += __shfl_xor(su, 16, 32);
    if (hi == 0)
        den_out[r0 + row] = S + rm + __builtin_amdgcn_logf(su) * LN2;
}

// ---------------------------------------------------------------------------
// Numerator: gold-path score. One thread per batch element; lane-consecutive b
// gives coalesced tag/mask loads.
// ---------------------------------------------------------------------------
__global__ __launch_bounds__(256)
void crf_score_kernel(const float* __restrict__ emissions,
                      const int*   __restrict__ tags,
                      const int*   __restrict__ mask,
                      const float* __restrict__ startT,
                      const float* __restrict__ endT,
                      const float* __restrict__ transitions,
                      float* __restrict__ num_out)
{
    const int b = blockIdx.x * blockDim.x + threadIdx.x;
    if (b >= BATCH) return;

    int   tg    = tags[b];
    float score = startT[tg] + emissions[(size_t)b * NUM_TAGS + tg];
    int   prev  = tg;
    int   msum  = mask[b];
    for (int t = 1; t < SEQ_LEN; ++t) {
        const int tgn = tags[t * BATCH + b];
        const int mk  = mask[t * BATCH + b];
        const float add = transitions[prev * NUM_TAGS + tgn]
                        + emissions[((size_t)t * BATCH + b) * NUM_TAGS + tgn];
        score = fmaf(add, (float)mk, score);
        prev = tgn;
        msum += mk;
    }
    const int lastTag = tags[(msum - 1) * BATCH + b];
    num_out[b] = score + endT[lastTag];
}

// ---------------------------------------------------------------------------
// Deterministic final reduction: sum(num - den) in a fixed order.
// ---------------------------------------------------------------------------
__global__ __launch_bounds__(32)
void crf_reduce_kernel(const float* __restrict__ num,
                       const float* __restrict__ den,
                       float* __restrict__ out)
{
    const int lane = threadIdx.x;
    float s = 0.0f;
    for (int i = lane; i < BATCH; i += 32) s += num[i] - den[i];
#pragma unroll
    for (int st = 16; st >= 1; st >>= 1) s += __shfl_xor(s, st, 32);
    if (lane == 0) out[0] = s;
}

extern "C" void kernel_launch(void* const* d_in, const int* in_sizes, int n_in,
                              void* d_out, int out_size, void* d_ws, size_t ws_size,
                              hipStream_t stream)
{
    const float* emissions = (const float*)d_in[0];   // (1024, 512, 64) f32
    const int*   tags      = (const int*)  d_in[1];   // (1024, 512) int
    const int*   mask      = (const int*)  d_in[2];   // (1024, 512) int
    const float* startT    = (const float*)d_in[3];   // (64,)
    const float* endT      = (const float*)d_in[4];   // (64,)
    const float* trans     = (const float*)d_in[5];   // (64, 64)

    float* ws  = (float*)d_ws;
    float* num = ws;            // [BATCH]
    float* den = ws + BATCH;    // [BATCH]

    crf_score_kernel<<<(BATCH + 255) / 256, 256, 0, stream>>>(
        emissions, tags, mask, startT, endT, trans, num);
    crf_normalizer_kernel<<<BATCH / 16, 32, 0, stream>>>(
        emissions, mask, startT, endT, trans, den);
    crf_reduce_kernel<<<1, 32, 0, stream>>>(num, den, (float*)d_out);
}
